// Conv2D_58849641890406
// MI455X (gfx1250) — 8-bit matrix operand order repaired and checked by reading, compiled, not run on hardware
//
#include <hip/hip_runtime.h>
#include <stdint.h>

typedef __attribute__((ext_vector_type(8))) int v8i;
typedef __attribute__((ext_vector_type(4))) int v4i;

#define F_   64
#define H_   256
#define W_   256
#define C_   64
#define HO_  254
#define WO_  254
#define JB   16
#define LSTR 257

__device__ __forceinline__ int pack4(v4i q) {
    unsigned ab = __builtin_amdgcn_perm((unsigned)q.y, (unsigned)q.x, 0x04000400u);
    unsigned cd = __builtin_amdgcn_perm((unsigned)q.w, (unsigned)q.z, 0x04000400u);
    return (int)__builtin_amdgcn_perm(cd, ab, 0x05040100u);
}

__global__ __launch_bounds__(256)
void conv_iu8_wmma(const int* __restrict__ x, const int* __restrict__ w,
                   const int* __restrict__ bias,
                   const int* __restrict__ qm_p, const int* __restrict__ ex_p,
                   const int* __restrict__ zp_p,
                   float* __restrict__ out)
{
    __shared__ int acc[JB * LSTR];

    const int f   = blockIdx.y;
    const int j0  = blockIdx.x * JB;
    const int tid = threadIdx.x;

    for (int i = tid; i < JB * LSTR; i += 256) acc[i] = 0;

    const int lane = tid & 31;
    const int wv   = tid >> 5;
    const int t    = lane & 15;
    const int h    = lane >> 4;

    v8i B = {};
    if (t < 9) {
        const int* tb = w + (f * 9 + t) * C_ + h * 16;
        #pragma unroll
        for (int v = 0; v < 4; ++v) B[v]     = pack4(*(const v4i*)(tb + 4 * v));
        #pragma unroll
        for (int v = 0; v < 4; ++v) B[4 + v] = pack4(*(const v4i*)(tb + 32 + 4 * v));
    }
    const int m = t / 3;
    const int n = t - 3 * m;

    __syncthreads();

    const int jhi = (j0 + JB < HO_) ? (j0 + JB) : HO_;
    const int nr  = ((j0 + JB + 2 < H_) ? (j0 + JB + 2) : H_) - j0;

    for (int rr = 0; rr < nr; ++rr) {
        const int r = j0 + rr;
        const int  j     = r - m;
        const bool rowOK = (t < 9) && (j >= j0) && (j < jhi);
        const int  rbase = (j - j0) * LSTR;
        const int* rowp  = x + ((long)(f * H_ + r) * W_ + t) * C_ + h * 16;

        #pragma unroll
        for (int cc = 0; cc < 2; ++cc) {
            const int q0 = (wv * 2 + cc) * 16;

            const int* ap = rowp + q0 * C_;
            v8i A;
            #pragma unroll
            for (int g = 0; g < 4; ++g) {
                v4i lo = *(const v4i*)(ap + 8 * g + 16 * (g >> 1));
                v4i hi = *(const v4i*)(ap + 8 * g + 16 * (g >> 1) + 4);
                A[2 * g]     = pack4(lo);
                A[2 * g + 1] = pack4(hi);
            }

            v8i Cz = {};
            v8i D = __builtin_amdgcn_wmma_i32_16x16x64_iu8(true, A, true, B, Cz,
                                                           false, false);

            const int cbase = q0 + 8 * h - n;
            const int base  = rbase + cbase;
            if (rowOK && ((unsigned)cbase <= (unsigned)(WO_ - 8))) {
                #pragma unroll
                for (int v = 0; v < 8; ++v) atomicAdd(&acc[base + v], D[v]);
            } else if (rowOK) {
                #pragma unroll
                for (int v = 0; v < 8; ++v) {
                    if ((unsigned)(cbase + v) < (unsigned)WO_)
                        atomicAdd(&acc[base + v], D[v]);
                }
            }
        }
    }
    __syncthreads();

    const int qm = *qm_p;
    const int ex = *ex_p;
    const int zp = *zp_p;
    const int rm = (qm < 2147418112) ? ((qm + (1 << 15)) >> 16) : 32767;
    const int ts = 15 - ex;
    const long long rnd = 1LL << (ts - 1);
    const int bf = bias[f];
    const int nrows = jhi - j0;
    for (int i = tid; i < nrows * WO_; i += 256) {
        const int jr = i / WO_;
        const int c  = i - jr * WO_;
        long long a64 = (long long)(acc[jr * LSTR + c] + bf) * (long long)rm + rnd;
        int res = (int)(a64 >> ts) + zp;
        res = res < -128 ? -128 : (res > 127 ? 127 : res);
        out[((long)f * HO_ + (j0 + jr)) * WO_ + c] = (float)res;
    }
}

extern "C" void kernel_launch(void* const* d_in, const int* in_sizes, int n_in,
                              void* d_out, int out_size, void* d_ws, size_t ws_size,
                              hipStream_t stream) {
    (void)in_sizes; (void)n_in; (void)d_ws; (void)ws_size; (void)out_size;
    const int* x    = (const int*)d_in[0];
    const int* w    = (const int*)d_in[1];
    const int* bias = (const int*)d_in[2];
    const int* qm   = (const int*)d_in[3];
    const int* ex   = (const int*)d_in[4];
    const int* zp   = (const int*)d_in[5];
    float* out = (float*)d_out;

    dim3 grid((HO_ + JB - 1) / JB, F_);
    conv_iu8_wmma<<<grid, 256, 0, stream>>>(x, w, bias, qm, ex, zp, out);
}
